// MixtureOfExperts_81630148428076
// MI455X (gfx1250) — matrix operand order repaired and checked by reading, not compiled, not run on hardware
//
#include <hip/hip_runtime.h>

// ---------------- problem constants ----------------
constexpr int kT  = 1024;   // tokens
constexpr int kH  = 1024;   // hidden
constexpr int kF2 = 2048;   // routed ffn dim
constexpr int kF3 = 3072;   // shared ffn dim
constexpr int kS  = 8;      // shared experts
constexpr int kE  = 32;     // routed experts
constexpr int kK  = 4;      // top-k
constexpr int kR  = 64;     // router rank
constexpr float kEps = 1e-6f;

typedef __attribute__((ext_vector_type(16))) _Float16 v16h;
typedef __attribute__((ext_vector_type(8)))  _Float16 v8h;
typedef __attribute__((ext_vector_type(8)))  float    v8f;

// ---------------- scratch layout (bytes) ----------------
constexpr size_t OFF_XH   = 0;
constexpr size_t OFF_XN   = OFF_XH + (size_t)kT * kH * 2;
constexpr size_t OFF_HB   = OFF_XN + (size_t)kT * kH * 2;
constexpr size_t OFF_R    = OFF_HB + (size_t)kT * kF3 * 2;
constexpr size_t OFF_COMB = OFF_R + (size_t)kT * 4;
constexpr size_t OFF_CNT  = OFF_COMB + (size_t)kT * kE * 4;

// ---------------- CDNA5 async-copy helpers ----------------
// GLOBAL_LOAD_ASYNC_TO_LDS_B128: 16B/lane global -> LDS, tracked by ASYNCcnt.
__device__ inline void async_copy16(unsigned lds_off, const void* gaddr) {
  asm volatile("global_load_async_to_lds_b128 %0, %1, off"
               :: "v"(lds_off), "v"(gaddr)
               : "memory");
}
__device__ inline void wait_async0() {
  asm volatile("s_wait_asynccnt 0x0" ::: "memory");
}
// low 32 bits of a generic pointer to __shared__ = wave-relative LDS byte addr
__device__ inline unsigned lds_addr(const void* p) {
  return (unsigned)(uintptr_t)p;
}

// ---------------- small prep kernels ----------------
__global__ void k_init(const float* __restrict__ x, float* __restrict__ out,
                       float* __restrict__ counts) {
  int i = blockIdx.x * blockDim.x + threadIdx.x;
  if (i < kT * kH) out[i] = x[i];           // residual base (shared mean keeps +x)
  if (i < kE) counts[i] = 0.f;
}

__global__ void k_rms(const float* __restrict__ x, float* __restrict__ rbuf) {
  __shared__ float red[256];
  int t = blockIdx.x;
  float s = 0.f;
  for (int h = threadIdx.x; h < kH; h += 256) {
    float v = x[(size_t)t * kH + h];
    s += v * v;
  }
  red[threadIdx.x] = s;
  __syncthreads();
  for (int w = 128; w > 0; w >>= 1) {
    if ((int)threadIdx.x < w) red[threadIdx.x] += red[threadIdx.x + w];
    __syncthreads();
  }
  if (threadIdx.x == 0) rbuf[t] = rsqrtf(red[0] / (float)kH + kEps);
}

__global__ void k_cvt(const float* __restrict__ x, _Float16* __restrict__ xh) {
  int i = blockIdx.x * blockDim.x + threadIdx.x;
  if (i < kT * kH) xh[i] = (_Float16)x[i];
}

__global__ void k_xn(const float* __restrict__ x, const float* __restrict__ rbuf,
                     const float* __restrict__ sn_s, _Float16* __restrict__ xnh) {
  int i = blockIdx.x * blockDim.x + threadIdx.x;
  if (i < kT * kH) {
    int t = i / kH, h = i % kH;
    xnh[i] = (_Float16)(x[i] * rbuf[t] * sn_s[h]);
  }
}

// ---------------- router ----------------
__global__ __launch_bounds__(256)
void k_router(const float* __restrict__ x, const float* __restrict__ rd,
              const float* __restrict__ ru, float* __restrict__ comb,
              float* __restrict__ counts) {
  const int t = blockIdx.x;
  __shared__ float part[4][kR];
  __shared__ float tmp[kR];
  __shared__ float logit[kE];
  __shared__ int   sel[kK];
  __shared__ float wsel[kK];
  const int tid = threadIdx.x;
  const int r = tid & 63, p = tid >> 6;
  float acc = 0.f;
  for (int h = p * 256; h < (p + 1) * 256; ++h)
    acc += x[(size_t)t * kH + h] * rd[(size_t)h * kR + r];
  part[p][r] = acc;
  __syncthreads();
  if (tid < kR) tmp[tid] = part[0][tid] + part[1][tid] + part[2][tid] + part[3][tid];
  __syncthreads();
  if (tid < kE) {
    float l = 0.f;
    for (int rr = 0; rr < kR; ++rr) l += tmp[rr] * ru[(size_t)rr * kE + tid];
    logit[tid] = l;
  }
  __syncthreads();
  if (tid == 0) {
    unsigned taken = 0;
    for (int k = 0; k < kK; ++k) {
      int best = 0; float bv = -1e30f;
      for (int e = 0; e < kE; ++e) {
        if (taken & (1u << e)) continue;
        if (logit[e] > bv) { bv = logit[e]; best = e; }
      }
      taken |= 1u << best; sel[k] = best; wsel[k] = bv;
    }
    float m = wsel[0], sum = 0.f;
    for (int k = 0; k < kK; ++k) { wsel[k] = __expf(wsel[k] - m); sum += wsel[k]; }
    for (int k = 0; k < kK; ++k) wsel[k] /= sum;
    for (int k = 0; k < kK; ++k) atomicAdd(&counts[sel[k]], 1.f);
  }
  __syncthreads();
  if (tid < kE) {
    float w = 0.f;
    for (int k = 0; k < kK; ++k) if (sel[k] == tid) w = wsel[k];
    comb[(size_t)t * kE + tid] = w;
  }
}

// ---------------- WMMA fragment helpers ----------------
// A (16x32 f16): lane half hh, element j -> k = hh*8 + j + (j>=8 ? 8 : 0)
__device__ inline v16h frag_a(const _Float16* base, int hh) {
  const v8h lo = *reinterpret_cast<const v8h*>(base + hh * 8);
  const v8h hi = *reinterpret_cast<const v8h*>(base + 16 + hh * 8);
  v16h a;
  for (int q = 0; q < 8; ++q) { a[q] = lo[q]; a[q + 8] = hi[q]; }
  return a;
}
// B (32x16 f16): lane half hh, element j -> k = hh*8 + j + (j>=8 ? 8 : 0)
__device__ inline v16h frag_b(const _Float16* base, int hh) {
  const v8h lo = *reinterpret_cast<const v8h*>(base + hh * 8);
  const v8h hi = *reinterpret_cast<const v8h*>(base + 16 + hh * 8);
  v16h b;
  for (int q = 0; q < 8; ++q) { b[q] = lo[q]; b[q + 8] = hi[q]; }
  return b;
}

// pack two fp32 -> one dword of two f16 and store to LDS (4B aligned)
__device__ inline void pack_store(_Float16* dst, float lo, float hi) {
  union { _Float16 h[2]; unsigned u; } u0;
  u0.h[0] = (_Float16)lo; u0.h[1] = (_Float16)hi;
  *reinterpret_cast<unsigned*>(dst) = u0.u;
}

// ---------------- GEMM 1: Hb = silu(A*W1) .* (A*W3) ----------------
// A [M,KH] f16 rowmajor, W1/W3 [KH,F] f32 rowmajor, Hb [M,F] f16.
__global__ __launch_bounds__(256)
void k_gemm_swiglu(const _Float16* __restrict__ A,
                   const float* __restrict__ W1, const float* __restrict__ W3,
                   _Float16* __restrict__ Hb, int KH, int F,
                   const float* __restrict__ comb, int expert) {
  const int m0 = blockIdx.y * 64;
  const int n0 = blockIdx.x * 64;
  __shared__ int s_active;
  if (comb) {
    if (threadIdx.x == 0) s_active = 0;
    __syncthreads();
    if (threadIdx.x < 64 && comb[(size_t)(m0 + threadIdx.x) * kE + expert] != 0.f)
      atomicOr(&s_active, 1);
    __syncthreads();
    if (!s_active) return;
  }
  __shared__ _Float16 As[64 * 40];
  __shared__ _Float16 W1s[64 * 40];
  __shared__ _Float16 W3s[64 * 40];
  const int tid = threadIdx.x;
  const int wave = tid >> 5, lane = tid & 31;
  const int hh = lane >> 4, idx = lane & 15;
  const int wm = wave & 3, wn = wave >> 2;
  const int arow = tid >> 2, acg = tid & 3;      // A stage: 64 rows x 4x8 f16
  const int wkp = (tid >> 4) * 2, wnt = tid & 15; // W stage: k pair x 4 n
  const unsigned aoff = lds_addr(&As[arow * 40 + acg * 8]);
  v8f accg[2], accv[2];
  for (int j = 0; j < 2; ++j)
    for (int q = 0; q < 8; ++q) { accg[j][q] = 0.f; accv[j][q] = 0.f; }

  for (int kk = 0; kk < KH; kk += 32) {
    // A tile: async DMA straight into LDS (already f16, pure copy)
    async_copy16(aoff, A + (size_t)(m0 + arow) * KH + kk + acg * 8);
    // W1/W3 tiles: fp32 -> f16, packed-transposed to [n][k]
    {
      const size_t rb = (size_t)(kk + wkp) * F + n0 + wnt * 4;
      const float4 a0 = *reinterpret_cast<const float4*>(W1 + rb);
      const float4 a1 = *reinterpret_cast<const float4*>(W1 + rb + F);
      const float4 b0 = *reinterpret_cast<const float4*>(W3 + rb);
      const float4 b1 = *reinterpret_cast<const float4*>(W3 + rb + F);
      const float a0v[4] = {a0.x, a0.y, a0.z, a0.w};
      const float a1v[4] = {a1.x, a1.y, a1.z, a1.w};
      const float b0v[4] = {b0.x, b0.y, b0.z, b0.w};
      const float b1v[4] = {b1.x, b1.y, b1.z, b1.w};
      for (int q = 0; q < 4; ++q) {
        pack_store(&W1s[(wnt * 4 + q) * 40 + wkp], a0v[q], a1v[q]);
        pack_store(&W3s[(wnt * 4 + q) * 40 + wkp], b0v[q], b1v[q]);
      }
      if (kk + 32 < KH) {
        __builtin_prefetch(W1 + rb + (size_t)32 * F, 0, 0);
        __builtin_prefetch(W3 + rb + (size_t)32 * F, 0, 0);
      }
    }
    wait_async0();
    __syncthreads();
    const v16h a = frag_a(&As[(wm * 16 + idx) * 40], hh);
    for (int j = 0; j < 2; ++j) {
      const int nb = (wn * 32 + j * 16 + idx) * 40;
      const v16h b1 = frag_b(&W1s[nb], hh);
      const v16h b3 = frag_b(&W3s[nb], hh);
      accg[j] = __builtin_amdgcn_wmma_f32_16x16x32_f16(false, a, false, b1, (short)0, accg[j], false, false);
      accv[j] = __builtin_amdgcn_wmma_f32_16x16x32_f16(false, a, false, b3, (short)0, accv[j], false, false);
    }
    __syncthreads();
  }
  // epilogue: h = silu(g) * v  -> f16
  for (int j = 0; j < 2; ++j) {
    const int ncol = n0 + wn * 32 + j * 16 + idx;
    for (int rq = 0; rq < 8; ++rq) {
      const int mrow = m0 + wm * 16 + hh * 8 + rq;
      float g = accg[j][rq], v = accv[j][rq];
      float hv = (g / (1.f + __expf(-g))) * v;
      Hb[(size_t)mrow * F + ncol] = (_Float16)hv;
    }
  }
}

// ---------------- GEMM 2: Out += scale(row) * (A*W2) ----------------
__global__ __launch_bounds__(256)
void k_gemm_acc(const _Float16* __restrict__ A, const float* __restrict__ W2,
                float* __restrict__ Out, int KF,
                const float* __restrict__ comb, int expert, float cscale) {
  const int m0 = blockIdx.y * 64;
  const int n0 = blockIdx.x * 64;
  __shared__ int s_active;
  if (comb) {
    if (threadIdx.x == 0) s_active = 0;
    __syncthreads();
    if (threadIdx.x < 64 && comb[(size_t)(m0 + threadIdx.x) * kE + expert] != 0.f)
      atomicOr(&s_active, 1);
    __syncthreads();
    if (!s_active) return;
  }
  __shared__ _Float16 As[64 * 40];
  __shared__ _Float16 Ws[64 * 40];
  const int tid = threadIdx.x;
  const int wave = tid >> 5, lane = tid & 31;
  const int hh = lane >> 4, idx = lane & 15;
  const int wm = wave & 3, wn = wave >> 2;
  const int arow = tid >> 2, acg = tid & 3;
  const int wkp = (tid >> 4) * 2, wnt = tid & 15;
  const unsigned aoff = lds_addr(&As[arow * 40 + acg * 8]);
  v8f acc[2];
  for (int j = 0; j < 2; ++j)
    for (int q = 0; q < 8; ++q) acc[j][q] = 0.f;

  for (int kk = 0; kk < KF; kk += 32) {
    async_copy16(aoff, A + (size_t)(m0 + arow) * KF + kk + acg * 8);
    {
      const size_t rb = (size_t)(kk + wkp) * kH + n0 + wnt * 4;
      const float4 a0 = *reinterpret_cast<const float4*>(W2 + rb);
      const float4 a1 = *reinterpret_cast<const float4*>(W2 + rb + kH);
      const float a0v[4] = {a0.x, a0.y, a0.z, a0.w};
      const float a1v[4] = {a1.x, a1.y, a1.z, a1.w};
      for (int q = 0; q < 4; ++q)
        pack_store(&Ws[(wnt * 4 + q) * 40 + wkp], a0v[q], a1v[q]);
      if (kk + 32 < KF)
        __builtin_prefetch(W2 + rb + (size_t)32 * kH, 0, 0);
    }
    wait_async0();
    __syncthreads();
    const v16h a = frag_a(&As[(wm * 16 + idx) * 40], hh);
    for (int j = 0; j < 2; ++j) {
      const int nb = (wn * 32 + j * 16 + idx) * 40;
      const v16h b = frag_b(&Ws[nb], hh);
      acc[j] = __builtin_amdgcn_wmma_f32_16x16x32_f16(false, a, false, b, (short)0, acc[j], false, false);
    }
    __syncthreads();
  }
  for (int j = 0; j < 2; ++j) {
    const int ncol = n0 + wn * 32 + j * 16 + idx;
    for (int rq = 0; rq < 8; ++rq) {
      const int mrow = m0 + wm * 16 + hh * 8 + rq;
      float scale = comb ? comb[(size_t)mrow * kE + expert] : cscale;
      Out[(size_t)mrow * kH + ncol] += scale * acc[j][rq];
    }
  }
}

// ---------------- load-balance loss ----------------
__global__ void k_loss(const float* __restrict__ counts, float* __restrict__ out_loss) {
  if (threadIdx.x == 0 && blockIdx.x == 0) {
    float s = 0.f;
    for (int e = 0; e < kE; ++e) s += counts[e];
    float m = s / (float)kE;
    float v = 0.f;
    for (int e = 0; e < kE; ++e) { float d = counts[e] - m; v += d * d; }
    out_loss[0] = v / (float)(kE - 1);
  }
}

// ---------------- driver ----------------
extern "C" void kernel_launch(void* const* d_in, const int* in_sizes, int n_in,
                              void* d_out, int out_size, void* d_ws, size_t ws_size,
                              hipStream_t stream) {
  (void)in_sizes; (void)n_in; (void)out_size; (void)ws_size;
  const float* x   = (const float*)d_in[0];
  const float* sn  = (const float*)d_in[1];
  const float* sw1 = (const float*)d_in[2];
  const float* sw2 = (const float*)d_in[3];
  const float* sw3 = (const float*)d_in[4];
  const float* rw1 = (const float*)d_in[5];
  const float* rw2 = (const float*)d_in[6];
  const float* rw3 = (const float*)d_in[7];
  const float* rd  = (const float*)d_in[8];
  const float* ru  = (const float*)d_in[9];
  float* out = (float*)d_out;

  char* ws = (char*)d_ws;
  _Float16* xh   = (_Float16*)(ws + OFF_XH);
  _Float16* xnh  = (_Float16*)(ws + OFF_XN);
  _Float16* hb   = (_Float16*)(ws + OFF_HB);
  float*    rbuf = (float*)(ws + OFF_R);
  float*    comb = (float*)(ws + OFF_COMB);
  float*    cnt  = (float*)(ws + OFF_CNT);

  const int nElem = kT * kH;
  k_init<<<(nElem + 255) / 256, 256, 0, stream>>>(x, out, cnt);
  k_rms<<<kT, 256, 0, stream>>>(x, rbuf);
  k_cvt<<<(nElem + 255) / 256, 256, 0, stream>>>(x, xh);
  k_router<<<kT, 256, 0, stream>>>(x, rd, ru, comb, cnt);

  // shared experts: mean over s of SwiGLU(RMSNorm(x)) added onto residual
  for (int s = 0; s < kS; ++s) {
    k_xn<<<(nElem + 255) / 256, 256, 0, stream>>>(x, rbuf, sn + (size_t)s * kH, xnh);
    k_gemm_swiglu<<<dim3(kF3 / 64, kT / 64), 256, 0, stream>>>(
        xnh, sw1 + (size_t)s * kH * kF3, sw3 + (size_t)s * kH * kF3,
        hb, kH, kF3, nullptr, 0);
    k_gemm_acc<<<dim3(kH / 64, kT / 64), 256, 0, stream>>>(
        hb, sw2 + (size_t)s * kF3 * kH, out, kF3, nullptr, 0, 1.f / (float)kS);
  }
  // routed experts: skip 64-token tiles where combine column is all zero
  for (int e = 0; e < kE; ++e) {
    k_gemm_swiglu<<<dim3(kF2 / 64, kT / 64), 256, 0, stream>>>(
        xh, rw1 + (size_t)e * kH * kF2, rw3 + (size_t)e * kH * kF2,
        hb, kH, kF2, comb, e);
    k_gemm_acc<<<dim3(kH / 64, kT / 64), 256, 0, stream>>>(
        hb, rw2 + (size_t)e * kF2 * kH, out, kF2, comb, e, 1.f);
  }
  k_loss<<<1, 32, 0, stream>>>(cnt, out + (size_t)kT * kH);
}
